// GlobalEncoder_59983513256107
// MI455X (gfx1250) — compile-verified, matrix operand order checked by reading, not run on hardware
//
#include <hip/hip_runtime.h>
#include <hip/hip_bf16.h>

// Problem constants (match reference setup_inputs)
#define NN 50000            // nodes
#define NE 800000           // edges (before self loops)
#define CD 128              // hidden dim
#define NH 2                // heads
#define HC 256              // NH*CD
#define NL 2                // layers
#define EP (NE + NN)        // edges incl. self loops
#define NEG_SLOPE 0.2f
#define BN_EPS 1e-5f
#define NODES_PER_BLK 50    // NN / 1000

typedef __attribute__((ext_vector_type(16))) __bf16          v16bf;
typedef __attribute__((ext_vector_type(8)))  float           v8f;
typedef __attribute__((ext_vector_type(8)))  unsigned short  u16x8;

union ABPack { u16x8 u[2]; v16bf v; };

__device__ __forceinline__ unsigned short f32_to_bf16(float f) {
  unsigned int u = __float_as_uint(f);
  u += 0x7FFFu + ((u >> 16) & 1u);   // round-to-nearest-even
  return (unsigned short)(u >> 16);
}

__device__ __forceinline__ void atomic_max_f32(float* addr, float val) {
  int* ai = (int*)addr;
  int old = __float_as_int(*addr);
  while (__int_as_float(old) < val) {
    int assumed = old;
    old = atomicCAS(ai, assumed, __float_as_int(val));
    if (old == assumed) break;
  }
}

// ---------------- utility ----------------
__global__ void k_fill(float* __restrict__ p, float v, int n) {
  int i = blockIdx.x * blockDim.x + threadIdx.x;
  if (i < n) p[i] = v;
}

// x (fp32) -> bf16 bits, row-major [NN][CD]
__global__ void k_cvt_x(const float* __restrict__ x, unsigned short* __restrict__ xb) {
  int i = blockIdx.x * blockDim.x + threadIdx.x;   // exact grid NN*CD
  xb[i] = f32_to_bf16(x[i]);
}

// W [CD][HC] fp32 -> transposed bf16 [HC][CD] so B columns are contiguous
__global__ void k_cvt_w(const float* __restrict__ W, unsigned short* __restrict__ wbt) {
  int i = blockIdx.x * blockDim.x + threadIdx.x;   // exact grid CD*HC
  int k = i / HC, n = i % HC;
  wbt[n * CD + k] = f32_to_bf16(W[i]);
}

// ---------------- WMMA GEMM: h[NN][HC] = x[NN][CD] @ W[CD][HC] ----------------
// One wave per 16x16 output tile; K loop = CD/32 = 4 steps of v_wmma_f32_16x16x32_bf16.
__global__ __launch_bounds__(256) void k_gemm(const unsigned short* __restrict__ xb,
                                              const unsigned short* __restrict__ wbt,
                                              float* __restrict__ h) {
  int wid  = blockIdx.x * 8 + (threadIdx.x >> 5);  // tile id, exact: 50000 tiles
  int lane = threadIdx.x & 31;
  int tileN = wid & 15;          // HC/16 = 16 column tiles
  int tileM = wid >> 4;          // NN/16 = 3125 row tiles

  int rowA = tileM * 16 + (lane & 15);
  int colB = tileN * 16 + (lane & 15);
  int kofs = (lane < 16) ? 0 : 8;   // ISA 16-bit A/B lane K-offset

  const unsigned short* aBase = xb  + (size_t)rowA * CD + kofs;
  const unsigned short* bBase = wbt + (size_t)colB * CD + kofs;

  v8f c = {};
#pragma unroll
  for (int ks = 0; ks < CD; ks += 32) {
    ABPack a, b;
    a.u[0] = *(const u16x8*)(aBase + ks);        // K = ks + kofs + 0..7
    a.u[1] = *(const u16x8*)(aBase + ks + 16);   // K = ks + kofs + 16..23
    b.u[0] = *(const u16x8*)(bBase + ks);
    b.u[1] = *(const u16x8*)(bBase + ks + 16);
    c = __builtin_amdgcn_wmma_f32_16x16x32_bf16(false, a.v, false, b.v,
                                                (short)0, c, false, false);
  }

  int mBase = tileM * 16 + ((lane < 16) ? 0 : 8);
  int nCol  = tileN * 16 + (lane & 15);
  float* hp = h + (size_t)mBase * HC + nCol;
#pragma unroll
  for (int r = 0; r < 8; ++r) hp[(size_t)r * HC] = c[r];
}

// ---------------- per-node attention logits ----------------
// one wave per (node, head): al = <h[n,head,:], a[head,:]>
__global__ __launch_bounds__(256) void k_attn(const float* __restrict__ h,
                                              const float* __restrict__ a_src,
                                              const float* __restrict__ a_dst,
                                              float* __restrict__ als,
                                              float* __restrict__ ald) {
  int wid  = blockIdx.x * 8 + (threadIdx.x >> 5);  // exact: NN*NH waves
  int lane = threadIdx.x & 31;
  int node = wid >> 1, head = wid & 1;
  const float* hp = h + (size_t)node * HC + head * CD;
  float s = 0.f, d = 0.f;
#pragma unroll
  for (int c = lane; c < CD; c += 32) {
    float v = hp[c];
    s += v * a_src[head * CD + c];
    d += v * a_dst[head * CD + c];
  }
#pragma unroll
  for (int off = 16; off; off >>= 1) { s += __shfl_xor(s, off); d += __shfl_xor(d, off); }
  if (lane == 0) { als[wid] = s; ald[wid] = d; }
}

// ---------------- edge pass 1: leaky-relu logits + segment max ----------------
__global__ void k_edge_max(const long long* __restrict__ ei,
                           const float* __restrict__ als, const float* __restrict__ ald,
                           float* __restrict__ ebuf, float* __restrict__ mbuf) {
  int t = blockIdx.x * blockDim.x + threadIdx.x;   // (edge*NH + head)
  if (t >= EP * NH) return;
  int e = t >> 1, head = t & 1;
  long long s, d;
  if (e < NE) { s = ei[e]; d = ei[NE + e]; } else { s = d = (long long)(e - NE); }
  float v = als[s * NH + head] + ald[d * NH + head];
  v = (v > 0.f) ? v : NEG_SLOPE * v;
  ebuf[t] = v;
  atomic_max_f32(&mbuf[d * NH + head], v);
}

// ---------------- edge pass 2: softmax denominator ----------------
__global__ void k_edge_den(const long long* __restrict__ ei,
                           const float* __restrict__ ebuf, const float* __restrict__ mbuf,
                           float* __restrict__ den) {
  int t = blockIdx.x * blockDim.x + threadIdx.x;
  if (t >= EP * NH) return;
  int e = t >> 1, head = t & 1;
  long long d = (e < NE) ? ei[NE + e] : (long long)(e - NE);
  atomicAdd(&den[d * NH + head], __expf(ebuf[t] - mbuf[d * NH + head]));
}

// ---------------- edge pass 3: weighted scatter-add aggregation ----------------
// one wave per (edge, head); each lane handles 4 channels (float4)
__global__ __launch_bounds__(256) void k_edge_agg(const long long* __restrict__ ei,
                                                  const float* __restrict__ ebuf,
                                                  const float* __restrict__ mbuf,
                                                  const float* __restrict__ den,
                                                  const float* __restrict__ h,
                                                  float* __restrict__ agg) {
  int wid  = blockIdx.x * 8 + (threadIdx.x >> 5);  // exact: EP*NH waves
  int lane = threadIdx.x & 31;
  int e = wid >> 1, head = wid & 1;
  long long s, d;
  if (e < NE) { s = ei[e]; d = ei[NE + e]; } else { s = d = (long long)(e - NE); }
  float alpha = __expf(ebuf[wid] - mbuf[d * NH + head]) / den[d * NH + head];
  const float4* hp = (const float4*)(h + (size_t)s * HC + head * CD);
  float4 hv = hp[lane];
  float* op = agg + (size_t)d * HC + head * CD + lane * 4;
  atomicAdd(op + 0, alpha * hv.x);
  atomicAdd(op + 1, alpha * hv.y);
  atomicAdd(op + 2, alpha * hv.z);
  atomicAdd(op + 3, alpha * hv.w);
}

// ---------------- head mean + bias, BN partial sums ----------------
__global__ void k_node_fin(const float* __restrict__ agg, const float* __restrict__ bias,
                           float* __restrict__ y, float* __restrict__ sum,
                           float* __restrict__ sumsq) {
  int c  = threadIdx.x;                 // 128 threads = channels
  int n0 = blockIdx.x * NODES_PER_BLK;
  float b = bias[c];
  float s = 0.f, s2 = 0.f;
  for (int i = 0; i < NODES_PER_BLK; ++i) {
    int n = n0 + i;
    float v = 0.5f * (agg[(size_t)n * HC + c] + agg[(size_t)n * HC + CD + c]) + b;
    y[(size_t)n * CD + c] = v;
    s += v; s2 += v * v;
  }
  atomicAdd(&sum[c], s);
  atomicAdd(&sumsq[c], s2);
}

__global__ void k_bn_stats(const float* __restrict__ sum, const float* __restrict__ sumsq,
                           float* __restrict__ mu, float* __restrict__ rstd) {
  int c = threadIdx.x;
  float m = sum[c] * (1.f / NN);
  float v = sumsq[c] * (1.f / NN) - m * m;
  mu[c] = m;
  rstd[c] = rsqrtf(v + BN_EPS);
}

// ---------------- BN apply + ELU + residual ----------------
__global__ void k_bn_elu_res(const float* __restrict__ y, const float* __restrict__ mu,
                             const float* __restrict__ rstd, const float* __restrict__ gamma,
                             const float* __restrict__ beta, const float* __restrict__ xres,
                             float* __restrict__ out) {
  int i = blockIdx.x * blockDim.x + threadIdx.x;   // exact grid NN*CD
  int c = i & (CD - 1);
  float v = gamma[c] * (y[i] - mu[c]) * rstd[c] + beta[c];
  v = (v > 0.f) ? v : expm1f(v);
  out[i] = v + xres[i];
}

extern "C" void kernel_launch(void* const* d_in, const int* in_sizes, int n_in,
                              void* d_out, int out_size, void* d_ws, size_t ws_size,
                              hipStream_t stream) {
  const float*     x0    = (const float*)d_in[0];
  const long long* ei    = (const long long*)d_in[1];   // int64 edge_index [2][NE]
  const float*     W     = (const float*)d_in[2];       // [NL][CD][HC]
  const float*     aS    = (const float*)d_in[3];       // [NL][NH][CD]
  const float*     aD    = (const float*)d_in[4];
  const float*     bias  = (const float*)d_in[5];       // [NL][CD]
  const float*     gamma = (const float*)d_in[6];
  const float*     beta  = (const float*)d_in[7];
  float*           out   = (float*)d_out;

  // carve workspace (256B aligned chunks)
  char* p = (char*)d_ws;
  auto carve = [&](size_t bytes) -> void* {
    void* r = (void*)p;
    p += (bytes + 255) & ~(size_t)255;
    return r;
  };
  float*          xbuf = (float*)carve((size_t)NN * CD * 4);   // inter-layer activations
  float*          h    = (float*)carve((size_t)NN * HC * 4);   // GEMM output (reused as y)
  float*          agg  = (float*)carve((size_t)NN * HC * 4);   // scatter-add accumulator
  unsigned short* xb   = (unsigned short*)carve((size_t)NN * CD * 2);
  unsigned short* wbt  = (unsigned short*)carve((size_t)CD * HC * 2);
  float*          als  = (float*)carve((size_t)NN * NH * 4);
  float*          ald  = (float*)carve((size_t)NN * NH * 4);
  float*          mbuf = (float*)carve((size_t)NN * NH * 4);
  float*          den  = (float*)carve((size_t)NN * NH * 4);
  float*          ebuf = (float*)carve((size_t)EP * NH * 4);
  float*          sum  = (float*)carve(CD * 4);
  float*          sumsq= (float*)carve(CD * 4);
  float*          mu   = (float*)carve(CD * 4);
  float*          rstd = (float*)carve(CD * 4);

  for (int l = 0; l < NL; ++l) {
    const float* xin  = (l == 0) ? x0 : xbuf;
    float*       xout = (l == NL - 1) ? out : xbuf;

    // init accumulators
    k_fill<<<(NN * HC + 255) / 256, 256, 0, stream>>>(agg, 0.f, NN * HC);
    k_fill<<<(NN * NH + 255) / 256, 256, 0, stream>>>(den, 0.f, NN * NH);
    k_fill<<<(NN * NH + 255) / 256, 256, 0, stream>>>(mbuf, -__builtin_inff(), NN * NH);
    k_fill<<<1, 256, 0, stream>>>(sum, 0.f, CD);
    k_fill<<<1, 256, 0, stream>>>(sumsq, 0.f, CD);

    // linear projection via WMMA
    k_cvt_x<<<NN * CD / 256, 256, 0, stream>>>(xin, xb);
    k_cvt_w<<<CD * HC / 256, 256, 0, stream>>>(W + (size_t)l * CD * HC, wbt);
    k_gemm<<<(NN / 16) * (HC / 16) / 8, 256, 0, stream>>>(xb, wbt, h);

    // attention
    k_attn<<<NN * NH / 8, 256, 0, stream>>>(h, aS + l * NH * CD, aD + l * NH * CD, als, ald);
    k_edge_max<<<(EP * NH + 255) / 256, 256, 0, stream>>>(ei, als, ald, ebuf, mbuf);
    k_edge_den<<<(EP * NH + 255) / 256, 256, 0, stream>>>(ei, ebuf, mbuf, den);
    k_edge_agg<<<EP * NH / 8, 256, 0, stream>>>(ei, ebuf, mbuf, den, h, agg);

    // head mean + bias + BN + ELU + residual (y reuses h buffer)
    k_node_fin<<<NN / NODES_PER_BLK, CD, 0, stream>>>(agg, bias + l * CD, h, sum, sumsq);
    k_bn_stats<<<1, CD, 0, stream>>>(sum, sumsq, mu, rstd);
    k_bn_elu_res<<<NN * CD / 256, 256, 0, stream>>>(h, mu, rstd, gamma + l * CD,
                                                    beta + l * CD, xin, xout);
  }
  (void)in_sizes; (void)n_in; (void)out_size; (void)ws_size;
}
